// RelativeDecoder_44641890074845
// MI455X (gfx1250) — matrix operand order repaired and checked by reading, not compiled, not run on hardware
//
#include <hip/hip_runtime.h>
#include <hip/hip_bf16.h>

#define FIN 128
#define F1  256
#define F2  64
#define NE  25
#define F3  (F2 * NE)   // 1600
#define MT  16
#define TPB 256
#define TWO_PI_F 6.283185307179586f

typedef _Float16 half_t;
typedef __attribute__((ext_vector_type(16))) _Float16 v16h;
typedef __attribute__((ext_vector_type(8)))  _Float16 v8h;
typedef __attribute__((ext_vector_type(8)))  float    v8f;

// Workspace layout (in half elements): transposed f16 weights
#define WT1_OFF 0
#define WT2_OFF (F1 * FIN)                 // 32768
#define WT3_OFF (WT2_OFF + F2 * F1)        // 49152
#define WT4_OFF (WT3_OFF + F3 * F2)        // 151552  (W4 padded to [16][64])
#define WS_HALF_TOTAL (WT4_OFF + 16 * F2)  // 152576 halves = 305152 bytes

#define WMMA_F16(A, B, C) \
  __builtin_amdgcn_wmma_f32_16x16x32_f16(false, (A), false, (B), (short)0, (C), false, false)

// ---------------------------------------------------------------------------
// Convert + transpose weights to f16 (WTi[n][k] = Wi[k][n]) once per launch.
// W4 is additionally zero-padded from 3 to 16 output columns.
// ---------------------------------------------------------------------------
__global__ void rd_prep_weights(const float* __restrict__ W1,
                                const float* __restrict__ W2,
                                const float* __restrict__ W3,
                                const float* __restrict__ W4,
                                half_t* __restrict__ wt) {
  int i = blockIdx.x * blockDim.x + threadIdx.x;
  if (i < WT2_OFF) {                       // W1: [FIN][F1] -> WT1 [F1][FIN]
    int n = i / FIN, k = i % FIN;
    wt[i] = (half_t)W1[k * F1 + n];
  } else if (i < WT3_OFF) {                // W2: [F1][F2] -> WT2 [F2][F1]
    int j = i - WT2_OFF;
    int n = j / F1, k = j % F1;
    wt[i] = (half_t)W2[k * F2 + n];
  } else if (i < WT4_OFF) {                // W3: [F2][F3] -> WT3 [F3][F2]
    int j = i - WT3_OFF;
    int n = j / F2, k = j % F2;
    wt[i] = (half_t)W3[k * F3 + n];
  } else if (i < WS_HALF_TOTAL) {          // W4: [F2][3] -> WT4 [16][F2], padded
    int j = i - WT4_OFF;
    int n = j / F2, k = j % F2;
    wt[i] = (n < 3) ? (half_t)W4[k * 3 + n] : (half_t)0.f;
  }
}

// A-fragment (16x32 f16): lane row m = L&15; halves[0..15] =
// K k0+hi*16.. contiguous (hi = lane>>4). One 32B LDS load.
__device__ __forceinline__ v16h lds_A(const half_t* base, int stride, int lane, int k0) {
  const int m = lane & 15, hi = lane >> 4;
  const half_t* p = base + m * stride + k0 + hi * 16;
  v16h a = *(const v16h*)p;
  return a;
}

// B-fragment (32x16 f16): lane n = L&15 column; 16 contiguous K halves
// (K = kt*32 + (L>>4)*16 ..). One aligned 32B global load from WT[n][.].
__device__ __forceinline__ v16h glb_B(const half_t* __restrict__ wt, int K,
                                      int nt, int kt, int lane) {
  const int n = lane & 15, kh = lane >> 4;
  return *(const v16h*)(wt + (size_t)(nt * 16 + n) * K + kt * 32 + kh * 16);
}

// D-fragment store (16x16 f32 C/D layout): VGPR j -> (m = j + 8*(lane>>4), n = lane&15)
__device__ __forceinline__ void store_D(half_t* base, int stride, int nbase, int lane,
                                        v8f c, const float* __restrict__ bias) {
  const int n = lane & 15, mb = (lane >> 4) * 8;
  const float bv = bias[nbase + n];
#pragma unroll
  for (int j = 0; j < 8; ++j) {
    float v = c[j] + bv;
    v = v > 0.f ? v : 0.f;                 // fused relu
    base[(mb + j) * stride + nbase + n] = (half_t)v;
  }
}

// ---------------------------------------------------------------------------
// Fused decoder: 16 points per block, all activations in LDS, WMMA GEMMs.
// ---------------------------------------------------------------------------
__global__ __launch_bounds__(TPB) void rd_decoder_kernel(
    const float* __restrict__ feats, const half_t* __restrict__ wt,
    const float* __restrict__ b1, const float* __restrict__ b2,
    const float* __restrict__ b3, const float* __restrict__ b4,
    float* __restrict__ out, int n_pts) {
  // Manual LDS carving with aliasing to stay under the static-LDS cap:
  //   [0      , 51200) H3 (f16 16x1600) — front 4 KB doubles as Xh in stage A
  //   [51200  , 59392) H1 (f16 16x256)  — aliased by FC4 (f32 400x4) in stage D
  //   [59392  , 61440) H2 (f16 16x64)
  //   [61440  , 62016) ROT (f32 16x9)
  __shared__ __align__(16) unsigned char SMEM[62016];
  half_t* H3  = (half_t*)SMEM;
  half_t* Xh  = (half_t*)SMEM;            // alias: dead before H3 is written
  half_t* H1  = (half_t*)(SMEM + 51200);
  float*  FC4 = (float*)(SMEM + 51200);   // alias: H1 dead before stage D
  half_t* H2  = (half_t*)(SMEM + 59392);
  float*  ROT = (float*)(SMEM + 61440);

  const int tid  = threadIdx.x;
  const int lane = tid & 31;
  const int wave = tid >> 5;
  const int row0 = blockIdx.x * MT;

  // ---- Load inputs: X (f32 -> f16); per-point rotation matrices.
#pragma unroll
  for (int e = tid; e < MT * FIN; e += TPB) {
    int m = e >> 7, c = e & 127;
    int rr = row0 + m; if (rr >= n_pts) rr = n_pts - 1;
    Xh[e] = (half_t)feats[(size_t)rr * (3 + FIN) + 3 + c];
  }
  if (tid < MT) {
    int rr = row0 + tid; if (rr >= n_pts) rr = n_pts - 1;
    const float* fr = feats + (size_t)rr * (3 + FIN);
    float a = fr[0] * TWO_PI_F, b = fr[1] * TWO_PI_F, g = fr[2] * TWO_PI_F;
    float ca = cosf(a), sa = sinf(a);
    float cb = cosf(b), sb = sinf(b);
    float cg = cosf(g), sg = sinf(g);
    float* R = &ROT[tid * 9];                  // R = Rz*Ry*Rx (row-major)
    R[0] = cg * cb; R[1] = cg * sb * sa - sg * ca; R[2] = cg * sb * ca + sg * sa;
    R[3] = sg * cb; R[4] = sg * sb * sa + cg * ca; R[5] = sg * sb * ca - cg * sa;
    R[6] = -sb;     R[7] = cb * sa;              R[8] = cb * ca;
  }
  __syncthreads();

  // ---- Stage A: H1 = relu(X @ W1 + b1)   [16x128]@[128x256], 2 N-tiles/wave
  {
    v16h a0 = lds_A(Xh, FIN, lane, 0);
    v16h a1 = lds_A(Xh, FIN, lane, 32);
    v16h a2 = lds_A(Xh, FIN, lane, 64);
    v16h a3 = lds_A(Xh, FIN, lane, 96);
    const half_t* wt1 = wt + WT1_OFF;
#pragma unroll
    for (int t = 0; t < 2; ++t) {
      const int nt = wave * 2 + t;
      v8f acc = {};
      acc = WMMA_F16(a0, glb_B(wt1, FIN, nt, 0, lane), acc);
      acc = WMMA_F16(a1, glb_B(wt1, FIN, nt, 1, lane), acc);
      acc = WMMA_F16(a2, glb_B(wt1, FIN, nt, 2, lane), acc);
      acc = WMMA_F16(a3, glb_B(wt1, FIN, nt, 3, lane), acc);
      store_D(H1, F1, nt * 16, lane, acc, b1);
    }
  }
  __syncthreads();

  // ---- Stage B: H2 = relu(H1 @ W2 + b2)  [16x256]@[256x64], waves 0..3
  if (wave < 4) {
    const half_t* wt2 = wt + WT2_OFF;
    v8f acc = {};
#pragma unroll
    for (int kt = 0; kt < 8; ++kt) {
      v16h a = lds_A(H1, F1, lane, kt * 32);
      acc = WMMA_F16(a, glb_B(wt2, F1, wave, kt, lane), acc);
    }
    store_D(H2, F2, wave * 16, lane, acc, b2);
  }
  __syncthreads();

  // ---- Stage C: H3 = relu(H2 @ W3 + b3)  [16x64]@[64x1600], 100 N-tiles
  {
    v16h a0 = lds_A(H2, F2, lane, 0);
    v16h a1 = lds_A(H2, F2, lane, 32);
    const half_t* wt3 = wt + WT3_OFF;
#pragma unroll 2
    for (int nt = wave; nt < F3 / 16; nt += 8) {
      v8f acc = {};
      acc = WMMA_F16(a0, glb_B(wt3, F2, nt, 0, lane), acc);
      acc = WMMA_F16(a1, glb_B(wt3, F2, nt, 1, lane), acc);
      store_D(H3, F3, nt * 16, lane, acc, b3);
    }
  }
  __syncthreads();

  // ---- Stage D: FC4 = H3r @ W4 + b4 (pre-tanh), on the matrix pipe.
  // Flat H3 is exactly row-major [400][64]: (m*25+k)*64 + c. 25 M-tiles, K=64.
  {
    const half_t* wt4 = wt + WT4_OFF;
    const v16h b0 = glb_B(wt4, F2, 0, 0, lane);
    const v16h b1f = glb_B(wt4, F2, 0, 1, lane);
    const int n = lane & 15, mb = (lane >> 4) * 8;
    const float b4n = (n < 3) ? b4[n] : 0.f;
    for (int t = wave; t < NE; t += 8) {
      const half_t* ar = H3 + t * 16 * F2;
      v16h a0 = lds_A(ar, F2, lane, 0);
      v16h a1 = lds_A(ar, F2, lane, 32);
      v8f acc = {};
      acc = WMMA_F16(a0, b0, acc);
      acc = WMMA_F16(a1, b1f, acc);
      if (n < 3) {
#pragma unroll
        for (int j = 0; j < 8; ++j)
          FC4[(t * 16 + mb + j) * 4 + n] = acc[j] + b4n;
      }
    }
  }
  __syncthreads();

  // ---- Stage E: tanh + per-point rotation, 400 rows/block at full occupancy.
  for (int r = tid; r < MT * NE; r += TPB) {
    const int m = r / NE, k = r % NE;
    const float f0 = tanhf(FC4[r * 4 + 0]);
    const float f1 = tanhf(FC4[r * 4 + 1]);
    const float f2 = tanhf(FC4[r * 4 + 2]);
    if (row0 + m < n_pts) {
      const float* R = &ROT[m * 9];
      const size_t ob = ((size_t)(row0 + m) * NE + k) * 3;
      out[ob + 0] = f0 * R[0] + f1 * R[3] + f2 * R[6];
      out[ob + 1] = f0 * R[1] + f1 * R[4] + f2 * R[7];
      out[ob + 2] = f0 * R[2] + f1 * R[5] + f2 * R[8];
    }
  }
}

extern "C" void kernel_launch(void* const* d_in, const int* in_sizes, int n_in,
                              void* d_out, int out_size, void* d_ws, size_t ws_size,
                              hipStream_t stream) {
  const float* feats = (const float*)d_in[0];
  const float* W1 = (const float*)d_in[1];
  const float* b1 = (const float*)d_in[2];
  const float* W2 = (const float*)d_in[3];
  const float* b2 = (const float*)d_in[4];
  const float* W3 = (const float*)d_in[5];
  const float* b3 = (const float*)d_in[6];
  const float* W4 = (const float*)d_in[7];
  const float* b4 = (const float*)d_in[8];
  float* out = (float*)d_out;
  half_t* wt = (half_t*)d_ws;

  const int n_pts = in_sizes[0] / (3 + FIN);   // 100000
  const int nblk  = (n_pts + MT - 1) / MT;     // 6250

  rd_prep_weights<<<(WS_HALF_TOTAL + 255) / 256, 256, 0, stream>>>(W1, W2, W3, W4, wt);
  rd_decoder_kernel<<<nblk, TPB, 0, stream>>>(feats, wt, b1, b2, b3, b4, out, n_pts);
}
